// BiMamba2_7919919694491
// MI455X (gfx1250) — matrix operand order repaired and checked by reading, not compiled, not run on hardware
//
#include <hip/hip_runtime.h>
#include <math.h>

// ---------------- model constants ----------------
#define BB      32
#define LC      16384
#define LL      2048
#define DMODEL  128
#define DINNER  256
#define DSTATE  64
#define HEADDIM 64
#define NHEADS  4
#define DCONV   4
#define XBCD    384      // DINNER + 2*DSTATE
#define DIP     644      // 2*DINNER + 2*DSTATE + NHEADS
#define MROWS   (BB*LL)  // 65536
#define GEMM_NT 4        // N-tiles per wave (register blocking)

typedef __attribute__((ext_vector_type(16))) __bf16          v16bf;
typedef __attribute__((ext_vector_type(8)))  float           v8f;
typedef __attribute__((ext_vector_type(4)))  unsigned        v4u;
typedef __attribute__((ext_vector_type(8)))  unsigned        v8u;

struct APair { v4u lo; v4u hi; };   // 32B: two 16B halves of a 16-bit WMMA fragment

__device__ __forceinline__ unsigned short f2bf(float f) {
  unsigned u = __builtin_bit_cast(unsigned, f);
  unsigned r = (u + 0x7FFFu + ((u >> 16) & 1u)) >> 16;   // RNE
  return (unsigned short)r;
}
__device__ __forceinline__ float siluf(float x) { return x / (1.f + __expf(-x)); }
__device__ __forceinline__ float geluf(float x) { return 0.5f * x * (1.f + erff(x * 0.70710678118f)); }

// ---------------- features: i,q,amp,phase,dphase ----------------
__global__ __launch_bounds__(256) void k_features(const float* __restrict__ iq, float* __restrict__ feat) {
  int idx = blockIdx.x * 256 + threadIdx.x;
  if (idx >= BB * LC) return;
  int b = idx / LC, t = idx - b * LC;
  float i = iq[(b * 2 + 0) * LC + t];
  float q = iq[(b * 2 + 1) * LC + t];
  float amp = sqrtf(i * i + q * q + 1e-12f);
  float ph = atan2f(q, i);
  float php = ph;
  if (t > 0) {
    float ip = iq[(b * 2 + 0) * LC + t - 1];
    float qp = iq[(b * 2 + 1) * LC + t - 1];
    php = atan2f(qp, ip);
  }
  size_t base = (size_t)b * 5 * LC + t;
  feat[base + 0 * LC] = i;
  feat[base + 1 * LC] = q;
  feat[base + 2 * LC] = amp;
  feat[base + 3 * LC] = ph;
  feat[base + 4 * LC] = ph - php;
}

// ---------------- conv(k=7,pad=3) + BN + GELU (c1/c2) ----------------
__global__ __launch_bounds__(256) void k_conv_bn_gelu(
    const float* __restrict__ x, const float* __restrict__ w, const float* __restrict__ bias,
    const float* __restrict__ g, const float* __restrict__ bb2, const float* __restrict__ mn,
    const float* __restrict__ vr, float* __restrict__ y, int Cin, int Cout) {
  int idx = blockIdx.x * 256 + threadIdx.x;
  int total = BB * Cout * LC;
  if (idx >= total) return;
  int t  = idx % LC;
  int co = (idx / LC) % Cout;
  int b  = idx / (LC * Cout);
  float acc = bias[co];
  for (int ci = 0; ci < Cin; ++ci) {
    const float* xr = x + ((size_t)b * Cin + ci) * LC;
    const float* wr = w + ((size_t)co * Cin + ci) * 7;
#pragma unroll
    for (int k = 0; k < 7; ++k) {
      int tt = t + k - 3;
      if (tt >= 0 && tt < LC) acc += xr[tt] * wr[k];
    }
  }
  float sc = g[co] * rsqrtf(vr[co] + 1e-5f);
  float xb = (acc - mn[co]) * sc + bb2[co];
  y[((size_t)b * Cout + co) * LC + t] = geluf(xb);
}

// ---------------- conv3 (k=8, stride 8) + BN + GELU, transposed bf16 out ----------------
__global__ __launch_bounds__(256) void k_conv3(
    const float* __restrict__ x, const float* __restrict__ w, const float* __restrict__ bias,
    const float* __restrict__ g, const float* __restrict__ bb2, const float* __restrict__ mn,
    const float* __restrict__ vr, unsigned short* __restrict__ y) {
  int idx = blockIdx.x * 256 + threadIdx.x;
  if (idx >= BB * LL * DMODEL) return;
  int co = idx % DMODEL;
  int l  = (idx / DMODEL) % LL;
  int b  = idx / (DMODEL * LL);
  float acc = bias[co];
  int t0 = l * 8;
  for (int ci = 0; ci < 64; ++ci) {
    const float* xr = x + ((size_t)b * 64 + ci) * LC + t0;
    const float* wr = w + ((size_t)co * 64 + ci) * 8;
#pragma unroll
    for (int k = 0; k < 8; ++k) acc += xr[k] * wr[k];
  }
  float sc = g[co] * rsqrtf(vr[co] + 1e-5f);
  float xb = (acc - mn[co]) * sc + bb2[co];
  y[((size_t)b * LL + l) * DMODEL + co] = f2bf(geluf(xb));
}

// ---------------- weight transpose + f32->bf16 : Wt[n*K+k] = W[k*N+n] ----------------
__global__ __launch_bounds__(256) void k_wt(const float* __restrict__ W, unsigned short* __restrict__ Wt,
                                            int K, int N) {
  int idx = blockIdx.x * 256 + threadIdx.x;
  if (idx >= K * N) return;
  int k = idx / N, n = idx - k * N;
  Wt[(size_t)n * K + k] = f2bf(W[(size_t)k * N + n]);
}

// ---------------- WMMA bf16 GEMM: C[M,N] = A[M,K] * Wt[N,K]^T ----------------
// Block: 8 waves = 8 M-tiles; blockIdx.y selects a 64-column N-panel staged to LDS
// via async global->LDS (ASYNCcnt). Each wave computes 16x64 (4 WMMA tiles, A reused).
// flipA: read A row b*L+(L-1-l); accFlip: C[b*L+(L-1-l)] += tile (else plain store).
__global__ __launch_bounds__(256) void k_gemm_wmma(
    const unsigned short* __restrict__ Abf, const unsigned short* __restrict__ Wt,
    float* __restrict__ C, int M, int N, int K, int Ld, int flipA, int accFlip) {
  __shared__ unsigned short sB[64 * 256];   // B panel: 64 n-rows x K (K<=256) bf16
  int tid  = threadIdx.x;
  int wave = tid >> 5, lane = tid & 31;
  int mloc = lane & 15, khalf = lane >> 4;
  int nbase = blockIdx.y * 64;

  // ---- stage B panel into LDS: memory -> LDS direct (no VGPR round trip) ----
  {
    const char* src = (const char*)(Wt + (size_t)nbase * K);
    unsigned lds0 = (unsigned)(uintptr_t)(&sB[0]);   // low 32 bits = LDS offset
    int bytes = 64 * K * 2;
    for (int off = tid * 16; off < bytes; off += 256 * 16) {
      asm volatile("global_load_async_to_lds_b128 %0, %1, %2"
                   :: "v"(lds0 + (unsigned)off), "v"(off), "s"(src)
                   : "memory");
    }
    asm volatile("s_wait_asynccnt 0x0" ::: "memory");
    __syncthreads();
  }

  int tM = blockIdx.x * 8 + wave;
  int m = tM * 16 + mloc;
  int arow = m;
  if (flipA) { int b = m / Ld, l = m - b * Ld; arow = b * Ld + (Ld - 1 - l); }
  const v4u* ap4 = (const v4u*)(Abf + (size_t)arow * K);   // A row, 16B granules
  __builtin_prefetch(ap4, 0, 0);
  const v8u* lb8 = (const v8u*)&sB[0];                     // LDS panel, 32B granules

  v8f acc[GEMM_NT] = {};
  for (int k0 = 0; k0 < K; k0 += 32) {
    // A fragment (contiguous k order): 16B granules at k0/8 + 2*khalf and +1
    int g = (k0 >> 3) + 2 * khalf;
    APair apr; apr.lo = ap4[g]; apr.hi = ap4[g + 1];
    v16bf a = __builtin_bit_cast(v16bf, apr);
    // preload all four B fragments (each = 8 contiguous dwords, 32B aligned, in LDS)
    v8u braw[GEMM_NT];
#pragma unroll
    for (int nt = 0; nt < GEMM_NT; ++nt) {
      size_t dw = (((size_t)(nt * 16 + mloc)) * K + (size_t)k0 + 16 * khalf) >> 1;
      braw[nt] = lb8[dw >> 3];
    }
#pragma unroll
    for (int nt = 0; nt < GEMM_NT; ++nt) {
      v16bf bm = __builtin_bit_cast(v16bf, braw[nt]);
      acc[nt] = __builtin_amdgcn_wmma_f32_16x16x32_bf16(false, a, false, bm, (short)0, acc[nt], false, false);
    }
  }
  // D layout: lane n-col = lane&15 ; VGPR r -> m = r + 8*(lane>>4)
#pragma unroll
  for (int nt = 0; nt < GEMM_NT; ++nt) {
    int n = nbase + nt * 16 + mloc;
    if (n < N) {
#pragma unroll
      for (int r = 0; r < 8; ++r) {
        int mm = tM * 16 + r + 8 * khalf;
        int orow = mm;
        if (accFlip) { int b = mm / Ld, l = mm - b * Ld; orow = b * Ld + (Ld - 1 - l); }
        float* dst = C + (size_t)orow * N + n;
        if (accFlip) *dst += acc[nt][r]; else *dst = acc[nt][r];
      }
    }
  }
}

// ---------------- depthwise causal conv (k=4) + SiLU over xBC slice ----------------
__global__ __launch_bounds__(256) void k_dwconv(const float* __restrict__ zx,
                                                const float* __restrict__ cw,
                                                const float* __restrict__ cb,
                                                float* __restrict__ xbc) {
  int idx = blockIdx.x * 256 + threadIdx.x;
  if (idx >= BB * LL * XBCD) return;
  int c = idx % XBCD;
  int l = (idx / XBCD) % LL;
  int b = idx / (XBCD * LL);
  float acc = cb[c];
#pragma unroll
  for (int k = 0; k < DCONV; ++k) {
    int ll2 = l + k - (DCONV - 1);
    if (ll2 >= 0) acc += zx[((size_t)b * LL + ll2) * DIP + DINNER + c] * cw[c * DCONV + k];
  }
  xbc[((size_t)b * LL + l) * XBCD + c] = siluf(acc);
}

// ---------------- dt = softplus(dt_raw + bias); dA = exp(dt * -exp(A_log)) ----------------
__global__ __launch_bounds__(256) void k_dt(const float* __restrict__ zx,
                                            const float* __restrict__ dt_bias,
                                            const float* __restrict__ A_log,
                                            float* __restrict__ dAo, float* __restrict__ dto) {
  int idx = blockIdx.x * 256 + threadIdx.x;
  if (idx >= BB * LL * NHEADS) return;
  int h = idx & (NHEADS - 1);
  size_t row = (size_t)(idx >> 2);
  float x = zx[row * DIP + (2 * DINNER + 2 * DSTATE) + h] + dt_bias[h];
  float dt = (x > 20.f) ? x : log1pf(__expf(x));
  float A = -__expf(A_log[h]);
  dAo[row * NHEADS + h] = __expf(dt * A);
  dto[row * NHEADS + h] = dt;
}

// ---------------- sequential SSM scan: one block per (b, head) ----------------
__global__ __launch_bounds__(256) void k_scan(const float* __restrict__ xbc,
                                              const float* __restrict__ dAa,
                                              const float* __restrict__ dtv,
                                              float* __restrict__ y) {
  int bh = blockIdx.x;
  int b = bh >> 2, h = bh & 3;
  int t = threadIdx.x;
  int p = t & 63;           // headdim index
  int n0 = (t >> 6) * 16;   // this thread's 16-state slice in dstate
  float st[16];
#pragma unroll
  for (int j = 0; j < 16; ++j) st[j] = 0.f;

  __shared__ float sB[DSTATE], sC[DSTATE], sX[HEADDIM], sP[256];

  for (int l = 0; l < LL; ++l) {
    size_t base = (size_t)b * LL + l;
    if (t < 64)       sB[t]        = xbc[base * XBCD + DINNER + t];
    else if (t < 128) sC[t - 64]   = xbc[base * XBCD + DINNER + DSTATE + (t - 64)];
    else if (t < 192) sX[t - 128]  = xbc[base * XBCD + h * HEADDIM + (t - 128)];
    __syncthreads();

    float a  = dAa[base * NHEADS + h];
    float dt = dtv[base * NHEADS + h];
    float dx = dt * sX[p];
    float part = 0.f;
#pragma unroll
    for (int j = 0; j < 16; ++j) {
      st[j] = st[j] * a + dx * sB[n0 + j];
      part += st[j] * sC[n0 + j];
    }
    sP[t] = part;
    __syncthreads();
    if (t < 64) y[(base * NHEADS + h) * HEADDIM + t] = sP[t] + sP[t + 64] + sP[t + 128] + sP[t + 192];
  }
}

// ---------------- y = (scan + D*xh) * silu(z); RMSNorm * w; -> bf16 ----------------
__global__ __launch_bounds__(256) void k_gate(const float* __restrict__ y,
                                              const float* __restrict__ xbc,
                                              const float* __restrict__ zx,
                                              const float* __restrict__ Dp,
                                              const float* __restrict__ normw,
                                              unsigned short* __restrict__ ybf) {
  size_t row = blockIdx.x;
  int d = threadIdx.x;         // 0..255 ; d = h*64+p
  int h = d >> 6;
  float xh = xbc[row * XBCD + d];
  float yv = y[row * DINNER + d] + Dp[h] * xh;
  float z  = zx[row * DIP + d];
  yv *= siluf(z);
  __shared__ float red[256];
  red[d] = yv * yv;
  __syncthreads();
  for (int s = 128; s > 0; s >>= 1) { if (d < s) red[d] += red[d + s]; __syncthreads(); }
  float inv = rsqrtf(red[0] / (float)DINNER + 1e-5f);
  ybf[row * DINNER + d] = f2bf(yv * inv * normw[d]);
}

// ---------------- FiLM + bit head (per row) ----------------
__global__ __launch_bounds__(128) void k_film_bits(const float* __restrict__ acc,
                                                   const float* __restrict__ snr_db,
                                                   const float* __restrict__ film_w,
                                                   const float* __restrict__ film_b,
                                                   const float* __restrict__ bit_w,
                                                   const float* __restrict__ bit_b,
                                                   float* __restrict__ bits) {
  size_t row = blockIdx.x;
  int d = threadIdx.x;
  int b = (int)(row / LL);
  float s = (snr_db[b] + 10.f) / 30.f;
  float gamma = s * film_w[d] + film_b[d];
  float beta  = s * film_w[DMODEL + d] + film_b[DMODEL + d];
  float hv = acc[row * DMODEL + d];
  hv = hv * (1.f + gamma) + beta;
  __shared__ float red[128];
  red[d] = hv * bit_w[d];
  __syncthreads();
  for (int s2 = 64; s2 > 0; s2 >>= 1) { if (d < s2) red[d] += red[d + s2]; __syncthreads(); }
  if (d == 0) bits[row] = red[0] + bit_b[0];
}

// ---------------- SNR head (deterministic mean over L, FiLM commutes) ----------------
__global__ __launch_bounds__(128) void k_snr(const float* __restrict__ acc,
                                             const float* __restrict__ snr_db,
                                             const float* __restrict__ film_w,
                                             const float* __restrict__ film_b,
                                             const float* __restrict__ snr_w,
                                             const float* __restrict__ snr_b,
                                             float* __restrict__ outp) {
  int b = blockIdx.x;
  int d = threadIdx.x;
  float s = (snr_db[b] + 10.f) / 30.f;
  float gamma = s * film_w[d] + film_b[d];
  float beta  = s * film_w[DMODEL + d] + film_b[DMODEL + d];
  float sum = 0.f;
  for (int l = 0; l < LL; ++l) sum += acc[((size_t)b * LL + l) * DMODEL + d];
  float hmean = (sum / (float)LL) * (1.f + gamma) + beta;
  __shared__ float red[128];
  red[d] = hmean * snr_w[d];
  __syncthreads();
  for (int s2 = 64; s2 > 0; s2 >>= 1) { if (d < s2) red[d] += red[d + s2]; __syncthreads(); }
  if (d == 0) outp[b] = red[0] + snr_b[0];
}

// =============================================================================
extern "C" void kernel_launch(void* const* d_in, const int* in_sizes, int n_in,
                              void* d_out, int out_size, void* d_ws, size_t ws_size,
                              hipStream_t stream) {
  (void)in_sizes; (void)n_in; (void)out_size; (void)ws_size;
  // ---- inputs (setup_inputs insertion order; params pytree depth-first in insertion order) ----
  const float* iq      = (const float*)d_in[0];
  const float* snr_db  = (const float*)d_in[1];
  const float* c1_w = (const float*)d_in[2];  const float* c1_b = (const float*)d_in[3];
  const float* bn1g = (const float*)d_in[4];  const float* bn1b = (const float*)d_in[5];
  const float* bn1m = (const float*)d_in[6];  const float* bn1v = (const float*)d_in[7];
  const float* c2_w = (const float*)d_in[8];  const float* c2_b = (const float*)d_in[9];
  const float* bn2g = (const float*)d_in[10]; const float* bn2b = (const float*)d_in[11];
  const float* bn2m = (const float*)d_in[12]; const float* bn2v = (const float*)d_in[13];
  const float* c3_w = (const float*)d_in[14]; const float* c3_b = (const float*)d_in[15];
  const float* bn3g = (const float*)d_in[16]; const float* bn3b = (const float*)d_in[17];
  const float* bn3m = (const float*)d_in[18]; const float* bn3v = (const float*)d_in[19];
  const float *in_proj[2], *conv_w[2], *conv_b[2], *dt_bias[2], *A_log[2], *Dvec[2], *norm_w[2], *out_proj[2];
  for (int dir = 0; dir < 2; ++dir) {
    int base = 20 + 8 * dir;
    in_proj[dir] = (const float*)d_in[base + 0];
    conv_w[dir]  = (const float*)d_in[base + 1];
    conv_b[dir]  = (const float*)d_in[base + 2];
    dt_bias[dir] = (const float*)d_in[base + 3];
    A_log[dir]   = (const float*)d_in[base + 4];
    Dvec[dir]    = (const float*)d_in[base + 5];
    norm_w[dir]  = (const float*)d_in[base + 6];
    out_proj[dir]= (const float*)d_in[base + 7];
  }
  const float* film_w = (const float*)d_in[36];
  const float* film_b = (const float*)d_in[37];
  const float* bit_w  = (const float*)d_in[38];
  const float* bit_b  = (const float*)d_in[39];
  const float* snr_w  = (const float*)d_in[40];
  const float* snr_b  = (const float*)d_in[41];
  float* outp = (float*)d_out;

  // ---- workspace layout (explicit time-based reuse) ----
  char* ws = (char*)d_ws;
  const size_t MB1 = 1ull << 20;
  unsigned short* x3bf = (unsigned short*)(ws + 0);          // [0,16MB) B*L*128 bf16, live through both GEMM passes
  float* feat = (float*)(ws + 16 * MB1);                     // 10.5MB, dead after conv1
  float* h1   = (float*)(ws + 32 * MB1);                     // 64MB, dead after conv2
  float* h2   = (float*)(ws + 96 * MB1);                     // 128MB, dead after conv3
  float* zx   = (float*)(ws + 16 * MB1);                     // 161MB per-direction (reuses feat/h1/h2 region)
  float* xbc  = (float*)(ws + 224 * MB1);                    // 96MB
  float* dAa  = (float*)(ws + 320 * MB1);                    // 1MB
  float* dtv  = (float*)(ws + 321 * MB1);                    // 1MB
  float* ybuf = (float*)(ws + 322 * MB1);                    // 64MB
  unsigned short* ybf = (unsigned short*)(ws + 386 * MB1);   // 32MB
  float* accb = (float*)(ws + 418 * MB1);                    // 32MB
  unsigned short* wtIn[2]  = { (unsigned short*)(ws + 450 * MB1),
                               (unsigned short*)(ws + 450 * MB1 + 165 * 1024) };
  unsigned short* wtOut[2] = { (unsigned short*)(ws + 450 * MB1 + 330 * 1024),
                               (unsigned short*)(ws + 450 * MB1 + 395 * 1024) };

  // ---- weight prep (transpose + bf16) ----
  for (int dir = 0; dir < 2; ++dir) {
    k_wt<<<(DMODEL * DIP + 255) / 256, 256, 0, stream>>>(in_proj[dir],  wtIn[dir],  DMODEL, DIP);
    k_wt<<<(DINNER * DMODEL + 255) / 256, 256, 0, stream>>>(out_proj[dir], wtOut[dir], DINNER, DMODEL);
  }

  // ---- frontend ----
  k_features<<<(BB * LC + 255) / 256, 256, 0, stream>>>(iq, feat);
  k_conv_bn_gelu<<<(BB * 32 * LC + 255) / 256, 256, 0, stream>>>(feat, c1_w, c1_b, bn1g, bn1b, bn1m, bn1v, h1, 5, 32);
  k_conv_bn_gelu<<<(BB * 64 * LC + 255) / 256, 256, 0, stream>>>(h1, c2_w, c2_b, bn2g, bn2b, bn2m, bn2v, h2, 32, 64);
  k_conv3<<<(BB * LL * DMODEL + 255) / 256, 256, 0, stream>>>(h2, c3_w, c3_b, bn3g, bn3b, bn3m, bn3v, x3bf);

  // ---- bidirectional Mamba2 (dir 1 reads flipped input, un-flips + accumulates on output) ----
  for (int dir = 0; dir < 2; ++dir) {
    dim3 gIn(MROWS / 128, (DIP + 63) / 64);                  // 512 x 11
    k_gemm_wmma<<<gIn, 256, 0, stream>>>(x3bf, wtIn[dir], zx, MROWS, DIP, DMODEL, LL, dir, 0);
    k_dwconv<<<(BB * LL * XBCD + 255) / 256, 256, 0, stream>>>(zx, conv_w[dir], conv_b[dir], xbc);
    k_dt<<<(BB * LL * NHEADS + 255) / 256, 256, 0, stream>>>(zx, dt_bias[dir], A_log[dir], dAa, dtv);
    k_scan<<<BB * NHEADS, 256, 0, stream>>>(xbc, dAa, dtv, ybuf);
    k_gate<<<MROWS, 256, 0, stream>>>(ybuf, xbc, zx, Dvec[dir], norm_w[dir], ybf);
    dim3 gOut(MROWS / 128, (DMODEL + 63) / 64);              // 512 x 2
    k_gemm_wmma<<<gOut, 256, 0, stream>>>(ybf, wtOut[dir], accb, MROWS, DMODEL, DINNER, LL, 0, dir);
  }

  // ---- heads ----
  k_film_bits<<<MROWS, 128, 0, stream>>>(accb, snr_db, film_w, film_b, bit_w, bit_b, outp);
  k_snr<<<BB, 128, 0, stream>>>(accb, snr_db, film_w, film_b, snr_w, snr_b, outp + MROWS);
}
